// ADTNSublayer_37323265802675
// MI455X (gfx1250) — matrix operand order repaired and checked by reading, not compiled, not run on hardware
//
#include <hip/hip_runtime.h>
#include <stdint.h>

// ---- problem constants (from reference) ----
#define NF      4096      // IN_F == OUT_F
#define NG      32        // number of block linears
#define GSZ     128       // GS == GO
#define M_TILE  128       // rows per workgroup
#define LDK     136       // padded LDS row stride in bf16 elements (128 + 8)

typedef __attribute__((ext_vector_type(16))) __bf16       v16bf;
typedef __attribute__((ext_vector_type(8)))  float        v8f;
typedef __attribute__((ext_vector_type(4)))  unsigned int v4u;

union Frag {            // 8 VGPRs = 16 bf16 per lane
    v4u   q[2];
    v16bf v;
};

__global__ __launch_bounds__(256)
void block_linear_permute_wmma(const float* __restrict__ x,
                               const long long* __restrict__ perm,
                               const float* __restrict__ W,
                               const float* __restrict__ bias,
                               float* __restrict__ out)
{
    extern __shared__ char smem[];
    __bf16* sXhi = (__bf16*)smem;                 // [M_TILE][LDK]
    __bf16* sXlo = sXhi + M_TILE * LDK;           // [M_TILE][LDK]
    __bf16* sWhi = sXlo + M_TILE * LDK;           // [GSZ][LDK]
    __bf16* sWlo = sWhi + GSZ    * LDK;           // [GSZ][LDK]

    const int g     = blockIdx.x;                 // fast-varying -> L2 reuse of row tile
    const int mbase = blockIdx.y * M_TILE;
    const int tid   = threadIdx.x;

    // ---------- stage: permuted gather of x tile + W_g, split fp32 -> bf16 hi/lo ----------
    const int k    = tid & 127;                   // fixed column slot per thread
    const int r0   = tid >> 7;                    // 0 or 1
    const int gcol = (int)perm[g * GSZ + k];      // permuted source column (int64 input)
    const float* __restrict__ Wg = W + (size_t)g * GSZ * GSZ;

    #pragma unroll 4
    for (int i = 0; i < 64; ++i) {
        const int row = r0 + 2 * i;               // covers 0..127
        // x gather (scattered 4B reads, served from L2-resident row tile)
        float xv = x[(size_t)(mbase + row) * NF + gcol];
        __bf16 xh = (__bf16)xv;
        __bf16 xl = (__bf16)(xv - (float)xh);
        sXhi[row * LDK + k] = xh;
        sXlo[row * LDK + k] = xl;
        // W_g row-major, fully coalesced
        float wv = Wg[row * GSZ + k];
        __bf16 wh = (__bf16)wv;
        __bf16 wl = (__bf16)(wv - (float)wh);
        sWhi[row * LDK + k] = wh;
        sWlo[row * LDK + k] = wl;
    }
    __syncthreads();

    // ---------- compute: each wave owns a 16-row strip x all 128 output cols ----------
    const int wave  = tid >> 5;
    const int lane  = tid & 31;
    const int lm    = lane & 15;
    const int hi    = (lane >= 16);
    const int c0    = hi ? 16 : 0;                // A-fragment K half
    const int cb    = hi ? 16 : 0;                // B-fragment K half
    const int mrow0 = wave * 16;

    v8f acc[8];
    #pragma unroll
    for (int nb = 0; nb < 8; ++nb) {              // init accumulators with bias (free add)
        float bv = bias[g * GSZ + nb * 16 + lm];
        v8f a;
        #pragma unroll
        for (int j = 0; j < 8; ++j) a[j] = bv;
        acc[nb] = a;
    }

    #pragma unroll
    for (int kk = 0; kk < 4; ++kk) {              // K = 128 in steps of 32
        Frag ah, al;
        {   // A: row m = mrow0+lm, chunks at k = kk*32 + {c0, c0+8}
            const __bf16* pa = sXhi + (mrow0 + lm) * LDK + kk * 32 + c0;
            ah.q[0] = *(const v4u*)(pa);
            ah.q[1] = *(const v4u*)(pa + 8);
            const __bf16* pl = sXlo + (mrow0 + lm) * LDK + kk * 32 + c0;
            al.q[0] = *(const v4u*)(pl);
            al.q[1] = *(const v4u*)(pl + 8);
        }
        #pragma unroll
        for (int nb = 0; nb < 8; ++nb) {
            Frag bh, bl;
            // B = W^T: column n of B == row n of W (row-major in LDS, contiguous K half)
            const __bf16* pb = sWhi + (nb * 16 + lm) * LDK + kk * 32 + cb;
            bh.q[0] = *(const v4u*)(pb);
            bh.q[1] = *(const v4u*)(pb + 8);
            const __bf16* pc = sWlo + (nb * 16 + lm) * LDK + kk * 32 + cb;
            bl.q[0] = *(const v4u*)(pc);
            bl.q[1] = *(const v4u*)(pc + 8);
            // fp32-accurate product via 3-term bf16 split: hh + hl + lh
            acc[nb] = __builtin_amdgcn_wmma_f32_16x16x32_bf16(
                          false, ah.v, false, bh.v, (short)0, acc[nb], false, false);
            acc[nb] = __builtin_amdgcn_wmma_f32_16x16x32_bf16(
                          false, ah.v, false, bl.v, (short)0, acc[nb], false, false);
            acc[nb] = __builtin_amdgcn_wmma_f32_16x16x32_bf16(
                          false, al.v, false, bh.v, (short)0, acc[nb], false, false);
        }
    }

    // ---------- store: C/D layout -> VGPR r: m = r + (lane>=16)*8, n = lane&15 ----------
    const int mlo = mbase + mrow0 + (hi ? 8 : 0);
    #pragma unroll
    for (int nb = 0; nb < 8; ++nb) {
        const int ncol = g * GSZ + nb * 16 + lm;
        #pragma unroll
        for (int r = 0; r < 8; ++r) {
            out[(size_t)(mlo + r) * NF + ncol] = acc[nb][r];
        }
    }
}

extern "C" void kernel_launch(void* const* d_in, const int* in_sizes, int n_in,
                              void* d_out, int out_size, void* d_ws, size_t ws_size,
                              hipStream_t stream)
{
    const float*     x    = (const float*)d_in[0];
    const long long* perm = (const long long*)d_in[1];   // int64 permutation
    const float*     W    = (const float*)d_in[2];
    const float*     b    = (const float*)d_in[3];
    float*           out  = (float*)d_out;

    const int mtotal = in_sizes[0] / NF;                 // B*S = 16384
    const size_t lds_bytes = (size_t)(2 * M_TILE * LDK + 2 * GSZ * LDK) * 2; // 139,264 B

    // CDNA5 WGP has 320KB LDS; raise the >64KB dynamic-LDS cap (non-stream op, capture-safe)
    (void)hipFuncSetAttribute((const void*)block_linear_permute_wmma,
                              hipFuncAttributeMaxDynamicSharedMemorySize,
                              (int)lds_bytes);

    dim3 grid(NG, mtotal / M_TILE);   // g fastest -> all 32 groups of a row tile adjacent in L2
    dim3 block(256);                  // 8 wave32 waves
    hipLaunchKernelGGL(block_linear_permute_wmma, grid, block, lds_bytes, stream,
                       x, perm, W, b, out);
}
